// TemporalMemory_14894946583486
// MI455X (gfx1250) — compile-verified, matrix operand order checked by reading, not run on hardware
//
#include <hip/hip_runtime.h>
#include <hip/hip_bf16.h>

#define COLUMNS   1024
#define CELLS     16
#define NUM_CELLS (COLUMNS * CELLS)   // 16384
#define SEGS      32
#define SYNS      128
#define ACT_THR   10
#define VOL_LR    0.1f

#define WAVES_PER_WG 4
#define WG_THREADS   (WAVES_PER_WG * 32)
#define MASK_WORDS   (NUM_CELLS / 32)  // 512

typedef __attribute__((ext_vector_type(8))) int v8i;

// -----------------------------------------------------------------------------
// Row-sums of a 32seg x 128syn byte matrix (0/1 bytes) via V_WMMA_I32_16X16X64_IU8.
// raw: 32x32 dwords in LDS; raw[s][l] holds bytes for synapses 4l..4l+3 of seg s.
// A tile = 16 rows (segs sg*16..+15) x 64 K. B = all-ones -> every column of D
// is the row sum. Returns v8i: lane<16 -> cnt[sg*16 + r], lane>=16 -> cnt[sg*16+8+r].
// -----------------------------------------------------------------------------
__device__ __forceinline__ v8i tm_rowsum16(const unsigned* raw, int sg, int lane) {
    const unsigned* base = raw + (sg * 16 + (lane & 15)) * 32;
    const int half = (lane & 16) ? 2 : 0;
    v8i ones;
#pragma unroll
    for (int i = 0; i < 8; ++i) ones[i] = 0x01010101;
    v8i acc = {};
#pragma unroll
    for (int kh = 0; kh < 2; ++kh) {
        const int off = kh * 16 + half;   // 8-bit A 16x64 VGPR layout (ISA 7.12.2)
        v8i a;
        a[0] = (int)base[off + 0];  a[1] = (int)base[off + 1];
        a[2] = (int)base[off + 4];  a[3] = (int)base[off + 5];
        a[4] = (int)base[off + 8];  a[5] = (int)base[off + 9];
        a[6] = (int)base[off + 12]; a[7] = (int)base[off + 13];
        acc = __builtin_amdgcn_wmma_i32_16x16x64_iu8(false, a, false, ones, acc,
                                                     false, false);
    }
    return acc;
}

// -----------------------------------------------------------------------------
// Kernel 0: pack prev_active (1 byte per cell, numpy bool layout) into a bitmask.
// -----------------------------------------------------------------------------
__global__ void tm_build_prev_mask(const unsigned char* __restrict__ prev,
                                   unsigned* __restrict__ prevMask) {
    const int t = blockIdx.x * blockDim.x + threadIdx.x;
    if (t < MASK_WORDS) {
        unsigned bits = 0u;
        const unsigned char* p = prev + t * 32;
#pragma unroll
        for (int j = 0; j < 32; ++j) bits |= (unsigned)(p[j] != 0) << j;
        prevMask[t] = bits;
    }
}

#define DO_SYN(CC, VV, QQ, SH) do {                                            \
    int idx_ = (CC);                                                           \
    idx_ = idx_ < 0 ? 0 : (idx_ > (NUM_CELLS - 1) ? (NUM_CELLS - 1) : idx_);   \
    unsigned pre_ = (prevMask[idx_ >> 5] >> (idx_ & 31)) & 1u;                 \
    unsigned cok_ = (unsigned)((VV) > 0.5f) | (unsigned)((QQ) > 0.5f);         \
    aW |= (cok_ & pre_) << (SH);                                               \
    pW |= pre_ << (SH);                                                        \
} while (0)

// -----------------------------------------------------------------------------
// Pass A: per-cell phase-1 predictive state + best matching segment.
// One wave per cell; streams conn/vol/cons once (b128 per lane per segment row).
// -----------------------------------------------------------------------------
__global__ void __launch_bounds__(WG_THREADS) tm_passA(
        const int* __restrict__ conn, const float* __restrict__ vol,
        const float* __restrict__ cons, const unsigned* __restrict__ prevMaskG,
        unsigned char* __restrict__ cellInfo) {
    __shared__ unsigned prevMask[MASK_WORDS];
    __shared__ unsigned rawA[WAVES_PER_WG][SEGS][32];  // connected & pre bytes
    __shared__ unsigned rawP[WAVES_PER_WG][SEGS][32];  // pre bytes (overlaps)

    const int tid = threadIdx.x;
    for (int i = tid; i < MASK_WORDS; i += WG_THREADS) prevMask[i] = prevMaskG[i];
    __syncthreads();

    const int w = tid >> 5, lane = tid & 31;
    const int cell = blockIdx.x * WAVES_PER_WG + w;

    const int4*   c4 = (const int4*)conn;
    const float4* v4 = (const float4*)vol;
    const float4* q4 = (const float4*)cons;
    const long cellBase = (long)cell * (SEGS * SYNS / 4);

    for (int s = 0; s < SEGS; ++s) {
        const long i4 = cellBase + s * (SYNS / 4) + lane;
        if (s + 4 < SEGS) {
            __builtin_prefetch(c4 + i4 + 4 * (SYNS / 4), 0, 1);
            __builtin_prefetch(v4 + i4 + 4 * (SYNS / 4), 0, 1);
            __builtin_prefetch(q4 + i4 + 4 * (SYNS / 4), 0, 1);
        }
        const int4   cw = c4[i4];
        const float4 vw = v4[i4];
        const float4 qw = q4[i4];
        unsigned aW = 0u, pW = 0u;
        DO_SYN(cw.x, vw.x, qw.x, 0);
        DO_SYN(cw.y, vw.y, qw.y, 8);
        DO_SYN(cw.z, vw.z, qw.z, 16);
        DO_SYN(cw.w, vw.w, qw.w, 24);
        rawA[w][s][lane] = aW;
        rawP[w][s][lane] = pW;
    }
    __syncthreads();

    bool segFire = false;
    unsigned bestKey = 0u;   // (overlap<<8) | (63-seg): max -> argmax w/ first-tie
#pragma unroll
    for (int sg = 0; sg < 2; ++sg) {
        v8i cnt = tm_rowsum16(&rawA[w][0][0], sg, lane);
        v8i ov  = tm_rowsum16(&rawP[w][0][0], sg, lane);
        const int mbase = sg * 16 + ((lane & 16) ? 8 : 0);
#pragma unroll
        for (int r = 0; r < 8; ++r) {
            segFire |= (cnt[r] >= ACT_THR);
            const unsigned key = ((unsigned)ov[r] << 8) | (unsigned)(63 - (mbase + r));
            bestKey = key > bestKey ? key : bestKey;
        }
    }
    const unsigned otherKey = (unsigned)__shfl_xor((int)bestKey, 16, 32);
    bestKey = otherKey > bestKey ? otherKey : bestKey;
    const bool anyFire = __ballot(segFire ? 1 : 0) != 0ull;
    if (lane == 0) {
        const int bestSeg = 63 - (int)(bestKey & 0xFFu);
        cellInfo[cell] = (unsigned char)((anyFire ? 1 : 0) | (bestSeg << 1));
    }
}

// -----------------------------------------------------------------------------
// Middle kernel (1 block, 1024 threads = one per column): burst logic,
// new_active floats + bitmask, n_act/n_pred/acc, learn_on flag.
// -----------------------------------------------------------------------------
__global__ void __launch_bounds__(1024) tm_mid(
        const int* __restrict__ x, const float* __restrict__ modp,
        const unsigned char* __restrict__ cellInfo,
        const unsigned* __restrict__ prevMask, unsigned* __restrict__ newMask,
        float* __restrict__ outActive, float* __restrict__ accOut,
        float* __restrict__ learnOut) {
    __shared__ int sNact, sNpred, sAnyPrev;
    const int tid = threadIdx.x;
    if (tid == 0) { sNact = 0; sNpred = 0; sAnyPrev = 0; }
    if (tid < MASK_WORDS) newMask[tid] = 0u;
    __threadfence_block();
    __syncthreads();

    if (tid < MASK_WORDS && prevMask[tid] != 0u) atomicOr(&sAnyPrev, 1);

    const int c = tid;                       // column
    const bool colAct = x[c] > 0;
    unsigned p = 0u;
#pragma unroll
    for (int ci = 0; ci < CELLS; ++ci)
        p |= (unsigned)(cellInfo[c * CELLS + ci] & 1) << ci;
    const bool hasPred = (p != 0u);
    const unsigned na = colAct ? (hasPred ? p : 0xFFFFu) : 0u;
#pragma unroll
    for (int ci = 0; ci < CELLS; ++ci)
        outActive[c * CELLS + ci] = ((na >> ci) & 1u) ? 1.0f : 0.0f;
    atomicOr(&newMask[c >> 1], na << ((c & 1) * 16));
    if (colAct) { atomicAdd(&sNact, 1); if (hasPred) atomicAdd(&sNpred, 1); }
    __syncthreads();

    if (tid == 0) {
        const int nact = sNact, npred = sNpred;
        accOut[0] = (nact > 0) ? ((float)npred / (float)(nact > 1 ? nact : 1)) : 1.0f;
        learnOut[0] = (modp[0] != 0.0f && sAnyPrev) ? 1.0f : 0.0f;
    }
}

#define DO_SYNB(CC, VV, QQ, SH) do {                                           \
    int idx_ = (CC);                                                           \
    idx_ = idx_ < 0 ? 0 : (idx_ > (NUM_CELLS - 1) ? (NUM_CELLS - 1) : idx_);   \
    unsigned preP_ = (prevMask[idx_ >> 5] >> (idx_ & 31)) & 1u;                 \
    unsigned preN_ = (newMask[idx_ >> 5] >> (idx_ & 31)) & 1u;                  \
    float v_ = (VV) + (preP_ ? addP : addN);                                   \
    unsigned cok_ = (unsigned)(v_ > 0.5f) | (unsigned)((QQ) > 0.5f);           \
    aW |= (cok_ & preN_) << (SH);                                              \
} while (0)

// -----------------------------------------------------------------------------
// Pass B: phase-3 predictive state with vol2 reconstructed on the fly
// (delta only on winner cell's best segment). Streams conn/vol/cons once more.
// -----------------------------------------------------------------------------
__global__ void __launch_bounds__(WG_THREADS) tm_passB(
        const int* __restrict__ conn, const float* __restrict__ vol,
        const float* __restrict__ cons, const int* __restrict__ x,
        const float* __restrict__ modp, const unsigned* __restrict__ prevMaskG,
        const unsigned* __restrict__ newMaskG,
        const unsigned char* __restrict__ cellInfo,
        const float* __restrict__ learnF, float* __restrict__ outPred) {
    __shared__ unsigned prevMask[MASK_WORDS];
    __shared__ unsigned newMask[MASK_WORDS];
    __shared__ unsigned rawA[WAVES_PER_WG][SEGS][32];

    const int tid = threadIdx.x;
    for (int i = tid; i < MASK_WORDS; i += WG_THREADS) {
        prevMask[i] = prevMaskG[i];
        newMask[i]  = newMaskG[i];
    }
    __syncthreads();

    const int w = tid >> 5, lane = tid & 31;
    const int cell = blockIdx.x * WAVES_PER_WG + w;

    const unsigned info = cellInfo[cell];
    const int bestSeg = (int)(info >> 1);
    const bool winner = ((newMask[cell >> 5] >> (cell & 31)) & 1u) != 0u;
    const float mod = modp[0];
    const bool doLearn = winner && (learnF[0] != 0.0f);
    const float dP = mod * VOL_LR, dN = -mod * VOL_LR;

    const int4*   c4 = (const int4*)conn;
    const float4* v4 = (const float4*)vol;
    const float4* q4 = (const float4*)cons;
    const long cellBase = (long)cell * (SEGS * SYNS / 4);

    for (int s = 0; s < SEGS; ++s) {
        const long i4 = cellBase + s * (SYNS / 4) + lane;
        if (s + 4 < SEGS) {
            __builtin_prefetch(c4 + i4 + 4 * (SYNS / 4), 0, 1);
            __builtin_prefetch(v4 + i4 + 4 * (SYNS / 4), 0, 1);
            __builtin_prefetch(q4 + i4 + 4 * (SYNS / 4), 0, 1);
        }
        const bool isL = doLearn && (s == bestSeg);
        const float addP = isL ? dP : 0.0f;
        const float addN = isL ? dN : 0.0f;
        const int4   cw = c4[i4];
        const float4 vw = v4[i4];
        const float4 qw = q4[i4];
        unsigned aW = 0u;
        DO_SYNB(cw.x, vw.x, qw.x, 0);
        DO_SYNB(cw.y, vw.y, qw.y, 8);
        DO_SYNB(cw.z, vw.z, qw.z, 16);
        DO_SYNB(cw.w, vw.w, qw.w, 24);
        rawA[w][s][lane] = aW;
    }
    __syncthreads();

    bool segFire = false;
#pragma unroll
    for (int sg = 0; sg < 2; ++sg) {
        v8i cnt = tm_rowsum16(&rawA[w][0][0], sg, lane);
#pragma unroll
        for (int r = 0; r < 8; ++r) segFire |= (cnt[r] >= ACT_THR);
    }
    const bool anyFire = __ballot(segFire ? 1 : 0) != 0ull;
    if (lane == 0) {
        const bool colAct = x[cell >> 4] > 0;
        outPred[cell] = (anyFire && colAct) ? 1.0f : 0.0f;
    }
}

// -----------------------------------------------------------------------------
// ws layout (bytes):
//   [0,     16384) uint8 cellInfo : bit0=pred1, bits1..5=best_seg
//   [16384, 18432) prevMask[512]
//   [18432, 20480) newMask[512]
//   [20480, 20484) learn_on flag (float)
// -----------------------------------------------------------------------------
extern "C" void kernel_launch(void* const* d_in, const int* in_sizes, int n_in,
                              void* d_out, int out_size, void* d_ws, size_t ws_size,
                              hipStream_t stream) {
    (void)in_sizes; (void)n_in; (void)out_size; (void)ws_size;
    const int*           x     = (const int*)d_in[0];
    const float*         modp  = (const float*)d_in[1];
    const unsigned char* prevA = (const unsigned char*)d_in[2];  // numpy bool = 1B
    const int*           conn  = (const int*)d_in[3];
    const float*         vol   = (const float*)d_in[4];
    const float*         cons  = (const float*)d_in[5];

    float* outActive = (float*)d_out;
    float* outPred   = outActive + NUM_CELLS;
    float* accOut    = outActive + 2 * NUM_CELLS;

    unsigned char* ws       = (unsigned char*)d_ws;
    unsigned char* cellInfo = ws;
    unsigned*      prevMask = (unsigned*)(ws + 16384);
    unsigned*      newMask  = (unsigned*)(ws + 18432);
    float*         learnF   = (float*)(ws + 20480);

    tm_build_prev_mask<<<dim3(2), dim3(256), 0, stream>>>(prevA, prevMask);

    tm_passA<<<dim3(NUM_CELLS / WAVES_PER_WG), dim3(WG_THREADS), 0, stream>>>(
        conn, vol, cons, prevMask, cellInfo);

    tm_mid<<<dim3(1), dim3(1024), 0, stream>>>(
        x, modp, cellInfo, prevMask, newMask, outActive, accOut, learnF);

    tm_passB<<<dim3(NUM_CELLS / WAVES_PER_WG), dim3(WG_THREADS), 0, stream>>>(
        conn, vol, cons, x, modp, prevMask, newMask, cellInfo, learnF, outPred);
}
